// View_attention_21852793602625
// MI455X (gfx1250) — matrix operand order repaired and checked by reading, compiled, not run on hardware
//
#include <hip/hip_runtime.h>
#include <hip/hip_bf16.h>

// ---------------------------------------------------------------------------
// Problem constants (from reference): B=16, L=2048, D=512, HID=64
// ---------------------------------------------------------------------------
#define BATCH 16
#define SEQ   2048
#define DIM   512
#define HIDP  64
#define MTOT  (BATCH * SEQ)          // 32768 tokens
#define SCALE 0.044194173824159216f  // 512^-0.5

typedef __attribute__((ext_vector_type(16))) __bf16 v16bf;
typedef __attribute__((ext_vector_type(8)))  float  v8f;

// ---------------------------------------------------------------------------
// Async global->LDS staging (CDNA5 GLOBAL_LOAD_ASYNC_TO_LDS_B128, ASYNCcnt).
// Probe round 2 revealed the builtin exists with param0 = int4 addrspace(1)*
// (clang prints AS1 as "__device__"). Cast via addrspace-qualified typedefs
// (lowers to addrspacecast: flat->global / flat->LDS-offset).
// ---------------------------------------------------------------------------
#if defined(__HIP_DEVICE_COMPILE__) &&                                   \
    __has_builtin(__builtin_amdgcn_global_load_async_to_lds_b128) &&     \
    __has_builtin(__builtin_amdgcn_s_wait_asynccnt)
#define USE_ASYNC_LDS 1
typedef int async_i32x4 __attribute__((ext_vector_type(4)));
typedef __attribute__((address_space(1))) void gvoid_as;
typedef __attribute__((address_space(3))) void lvoid_as;
typedef __attribute__((address_space(1))) async_i32x4 g_i32x4;
typedef __attribute__((address_space(3))) async_i32x4 l_i32x4;
#else
#define USE_ASYNC_LDS 0
#endif

static __device__ __forceinline__ void copy16_g2l(const void* g, void* l) {
#if USE_ASYNC_LDS
  __builtin_amdgcn_global_load_async_to_lds_b128(
      (g_i32x4*)(gvoid_as*)const_cast<void*>(g),
      (l_i32x4*)(lvoid_as*)l, 0, 0);
#else
  *(uint4*)l = *(const uint4*)g;
#endif
}
static __device__ __forceinline__ void async_wait_all() {
#if USE_ASYNC_LDS
  __builtin_amdgcn_s_wait_asynccnt(0);
#endif
}

// ---------------------------------------------------------------------------
// WMMA fragment loaders. CDNA5 16-bit A-matrix 16x32 layout:
//   lane l (<16): M=l,    K = {0..7, 16..23}
//   lane l (>=16):M=l-16, K = {8..15, 24..31}
// B-matrix 32x16 layout (same K order as A):
//   lane l: N=l%16, K = {8*(l/16)..+7, 16+8*(l/16)..+7}
// Both reduce to two 16-byte contiguous loads per lane when the source is
// K-contiguous.
// ---------------------------------------------------------------------------
static __device__ __forceinline__ v16bf fragA_bf16(const __bf16* row, int half) {
  const int off = half * 8;
  union { uint4 u[2]; v16bf v; } t;
  t.u[0] = *reinterpret_cast<const uint4*>(row + off);
  t.u[1] = *reinterpret_cast<const uint4*>(row + off + 16);
  return t.v;
}
static __device__ __forceinline__ v16bf fragB_bf16(const __bf16* row, int half) {
  const int off = half * 8;
  union { uint4 u[2]; v16bf v; } t;
  t.u[0] = *reinterpret_cast<const uint4*>(row + off);
  t.u[1] = *reinterpret_cast<const uint4*>(row + off + 16);
  return t.v;
}
static __device__ __forceinline__ v16bf fragA_f32(const float* row, int half) {
  const int off = half * 8;
  v16bf v;
#pragma unroll
  for (int i = 0; i < 8; ++i) v[i] = (__bf16)row[off + i];
#pragma unroll
  for (int i = 0; i < 8; ++i) v[8 + i] = (__bf16)row[off + 16 + i];
  return v;
}

static __device__ __forceinline__ v8f wmma_bf16(v16bf a, v16bf b, v8f c) {
  return __builtin_amdgcn_wmma_f32_16x16x32_bf16(false, a, false, b, (short)0, c,
                                                 false, false);
}

// ---------------------------------------------------------------------------
// Kernel 1: transpose+convert the four 512x512 fp32 weights to bf16 Wt[N,K]
// ---------------------------------------------------------------------------
__global__ void __launch_bounds__(256)
prep_w(const float* __restrict__ Wq, const float* __restrict__ Wk,
       const float* __restrict__ Wv, const float* __restrict__ Wo,
       __bf16* __restrict__ tq, __bf16* __restrict__ tk,
       __bf16* __restrict__ tv, __bf16* __restrict__ to_) {
  int idx = blockIdx.x * 256 + threadIdx.x;        // 4 * 512 * 512 total
  int w   = idx >> 18;
  int rem = idx & ((1 << 18) - 1);
  int k   = rem >> 9;
  int n   = rem & 511;
  const float* src = (w == 0) ? Wq : (w == 1) ? Wk : (w == 2) ? Wv : Wo;
  __bf16*      dst = (w == 0) ? tq : (w == 1) ? tk : (w == 2) ? tv : to_;
  dst[(size_t)n * DIM + k] = (__bf16)src[(size_t)k * DIM + n];
}

// ---------------------------------------------------------------------------
// Kernel 2: per-token LayerNorm(q) -> bf16 qn, and pos-MLP -> bf16 pe
// ---------------------------------------------------------------------------
__global__ void __launch_bounds__(128)
prep_token(const float* __restrict__ q, const float* __restrict__ pos,
           const float* __restrict__ g, const float* __restrict__ beta,
           const float* __restrict__ pw1, const float* __restrict__ pb1,
           const float* __restrict__ pw2, const float* __restrict__ pb2,
           __bf16* __restrict__ qn, __bf16* __restrict__ pe) {
  __shared__ float red[128];
  __shared__ float h[HIDP];
  const int tok = blockIdx.x;
  const int tid = threadIdx.x;
  const int d0  = tid * 4;

  float4 x = *reinterpret_cast<const float4*>(q + (size_t)tok * DIM + d0);

  red[tid] = x.x + x.y + x.z + x.w;
  __syncthreads();
  for (int s = 64; s > 0; s >>= 1) {
    if (tid < s) red[tid] += red[tid + s];
    __syncthreads();
  }
  const float mu = red[0] * (1.0f / DIM);
  __syncthreads();

  float4 xc = {x.x - mu, x.y - mu, x.z - mu, x.w - mu};
  red[tid] = xc.x * xc.x + xc.y * xc.y + xc.z * xc.z + xc.w * xc.w;
  __syncthreads();
  for (int s = 64; s > 0; s >>= 1) {
    if (tid < s) red[tid] += red[tid + s];
    __syncthreads();
  }
  const float rstd = rsqrtf(red[0] * (1.0f / DIM) + 1e-6f);

  {
    float c[4] = {xc.x, xc.y, xc.z, xc.w};
#pragma unroll
    for (int i = 0; i < 4; ++i)
      qn[(size_t)tok * DIM + d0 + i] =
          (__bf16)(c[i] * rstd * g[d0 + i] + beta[d0 + i]);
  }

  if (tid < HIDP) {
    const float* pp = pos + (size_t)tok * 4;
    float s = pb1[tid];
#pragma unroll
    for (int j = 0; j < 4; ++j) s += pp[j] * pw1[j * HIDP + tid];
    h[tid] = fmaxf(s, 0.0f);
  }
  __syncthreads();

  float acc[4];
#pragma unroll
  for (int i = 0; i < 4; ++i) acc[i] = pb2[d0 + i];
  for (int j = 0; j < HIDP; ++j) {
    const float hj = h[j];
#pragma unroll
    for (int i = 0; i < 4; ++i) acc[i] += hj * pw2[(size_t)j * DIM + d0 + i];
  }
#pragma unroll
  for (int i = 0; i < 4; ++i) pe[(size_t)tok * DIM + d0 + i] = (__bf16)acc[i];
}

// ---------------------------------------------------------------------------
// Kernel 3: projection GEMM  C[M,512] = A[M,512] @ W  (Wt is [N,K] bf16).
// Compile-time variants (branch-free inner loops):
//   AF32: A operand fp32 (converted in loader) vs bf16
//   ADDPE: fuse +pe epilogue; TRANS: store bf16 transposed [N, M]
// Block = 256 thr = 8 waves -> 64(M) x 128(N); wave = 16x64 (4 wmma N-tiles).
// ---------------------------------------------------------------------------
template <int AF32, int ADDPE, int TRANS>
__global__ void __launch_bounds__(256)
gemm512(const __bf16* __restrict__ Abf, const float* __restrict__ Af32,
        const __bf16* __restrict__ Wt, const __bf16* __restrict__ peb,
        __bf16* __restrict__ outb, __bf16* __restrict__ outT) {
  const int lane = threadIdx.x & 31, wid = threadIdx.x >> 5;
  const int half = lane >> 4, ln = lane & 15;
  const int m0 = blockIdx.x * 64 + (wid >> 1) * 16;
  const int n0 = blockIdx.y * 128 + (wid & 1) * 64;

  v8f acc[4];
#pragma unroll
  for (int t = 0; t < 4; ++t) acc[t] = (v8f){0, 0, 0, 0, 0, 0, 0, 0};

  const size_t arow = (size_t)(m0 + ln) * DIM;
  for (int ks = 0; ks < 16; ++ks) {
    const int kk = ks * 32;
    v16bf a;
    if constexpr (AF32) a = fragA_f32(Af32 + arow + kk, half);
    else                a = fragA_bf16(Abf + arow + kk, half);
#pragma unroll
    for (int t = 0; t < 4; ++t) {
      const __bf16* wrow = Wt + (size_t)(n0 + t * 16 + ln) * DIM + kk;
      acc[t] = wmma_bf16(a, fragB_bf16(wrow, half), acc[t]);
    }
  }

#pragma unroll
  for (int t = 0; t < 4; ++t) {
    const int n = n0 + t * 16 + ln;
#pragma unroll
    for (int r = 0; r < 8; ++r) {
      const int m = m0 + r + half * 8;
      float v = acc[t][r];
      if constexpr (ADDPE) v += (float)peb[(size_t)m * DIM + n];
      if constexpr (TRANS) outT[(size_t)n * MTOT + m] = (__bf16)v;
      else                 outb[(size_t)m * DIM + n] = (__bf16)v;
    }
  }
}

// ---------------------------------------------------------------------------
// Kernel 4: causal flash attention with double-buffered async LDS staging.
// Grid: (L/64, B). Block = 256 thr = 8 waves = 4 q-subtiles x 2 D-halves.
// Per key block of 32: stage K[32,512] and V^T[512,32] tiles into LDS via
// GLOBAL_LOAD_ASYNC_TO_LDS_B128 (prefetching tile kb+1 while WMMAs run on kb),
// compute S via wmma, analytic causal mask, online softmax (lane shuffles),
// relay P C->A layout through per-wave LDS, accumulate O via wmma.
// All waves run to the block-wide key bound; per-wave skip is uniform so
// EXEC stays all-ones around WMMA.
// ---------------------------------------------------------------------------
#define KROW 520   // kbuf row stride (bf16): 1040B = 16B*65, dword stride 260
#define VROW 40    // vbuf row stride (bf16): 80B  = 16B*5,  dword stride 20
#define PROW 40    // plds row stride

__global__ void __launch_bounds__(256)
flash_attn(const __bf16* __restrict__ qp, const __bf16* __restrict__ kp,
           const __bf16* __restrict__ vpet, __bf16* __restrict__ ao) {
  __shared__ __align__(16) __bf16 kbuf[2][32][KROW];    // 2 x 32.5 KB
  __shared__ __align__(16) __bf16 vbuf[2][DIM][VROW];   // 2 x 40 KB
  __shared__ __align__(16) __bf16 plds[8][16][PROW];    // 10 KB

  const int tid  = threadIdx.x;
  const int lane = tid & 31, wid = tid >> 5;
  const int half = lane >> 4, ln = lane & 15;
  const int qsub = wid >> 1;          // 0..3 : which 16-row q tile
  const int dh   = wid & 1;           // 0..1 : which 256-wide D half
  const int bb   = blockIdx.y;
  const int qb0  = blockIdx.x * 64;
  const int qtb  = qb0 + qsub * 16;   // query row base in batch
  const int myKbMax  = (qtb + 15) >> 5;
  const int blkKbMax = (qb0 + 63) >> 5;

  v8f o[16];
#pragma unroll
  for (int t = 0; t < 16; ++t) o[t] = (v8f){0, 0, 0, 0, 0, 0, 0, 0};
  float mrow[8], lrow[8];
#pragma unroll
  for (int r = 0; r < 8; ++r) { mrow[r] = -3.0e38f; lrow[r] = 0.0f; }

  const __bf16* qrow = qp + ((size_t)bb * SEQ + qtb + ln) * DIM;

  // cooperative staging of one 32-key tile (K: 32x1KB, V^T: 512x64B)
  auto stage = [&](int buf, int kb) {
    const int k0 = kb * 32;
    const char* ksrc = (const char*)(kp + ((size_t)bb * SEQ + k0) * DIM);
#pragma unroll
    for (int i = 0; i < 8; ++i) {          // 2048 x 16B chunks, 8 per thread
      const int c = tid + i * 256;
      const int row = c >> 6, col = c & 63;
      copy16_g2l(ksrc + (size_t)row * (DIM * 2) + col * 16,
                 (char*)&kbuf[buf][row][0] + col * 16);
    }
    const char* vsrc = (const char*)(vpet + (size_t)bb * SEQ + k0);
#pragma unroll
    for (int i = 0; i < 8; ++i) {          // 2048 x 16B chunks, 8 per thread
      const int c = tid + i * 256;
      const int row = c >> 2, col = c & 3;
      copy16_g2l(vsrc + (size_t)row * (MTOT * 2) + col * 16,
                 (char*)&vbuf[buf][row][0] + col * 16);
    }
  };

  stage(0, 0);
  async_wait_all();
  __syncthreads();

  for (int kb = 0; kb <= blkKbMax; ++kb) {
    const int cur = kb & 1;
    if (kb < blkKbMax) stage(cur ^ 1, kb + 1);  // async prefetch next tile

    if (kb <= myKbMax) {                        // uniform per wave
      const int k0 = kb * 32;
      v8f s0 = (v8f){0, 0, 0, 0, 0, 0, 0, 0};
      v8f s1 = (v8f){0, 0, 0, 0, 0, 0, 0, 0};

      // S[16,32] = Q[16,512] @ K^T : contraction over d in 16 steps of 32
      for (int ks = 0; ks < 16; ++ks) {
        const int kk = ks * 32;
        v16bf a = fragA_bf16(qrow + kk, half);
        s0 = wmma_bf16(a, fragB_bf16(&kbuf[cur][ln][kk], half), s0);
        s1 = wmma_bf16(a, fragB_bf16(&kbuf[cur][ln + 16][kk], half), s1);
      }

      // online softmax (row r: m = qtb + r + 8*half, keys across lanes)
      const int nk0 = k0 + ln;
#pragma unroll
      for (int r = 0; r < 8; ++r) {
        const int m = qtb + r + half * 8;
        float v0 = s0[r] * SCALE; if (nk0 > m)      v0 = -3.0e38f;
        float v1 = s1[r] * SCALE; if (nk0 + 16 > m) v1 = -3.0e38f;

        float t = fmaxf(v0, v1);
        t = fmaxf(t, __shfl_xor(t, 8, 32));
        t = fmaxf(t, __shfl_xor(t, 4, 32));
        t = fmaxf(t, __shfl_xor(t, 2, 32));
        t = fmaxf(t, __shfl_xor(t, 1, 32));

        const float mnew = fmaxf(mrow[r], t);
        const float corr = __expf(mrow[r] - mnew);
        const float p0 = __expf(v0 - mnew);
        const float p1 = __expf(v1 - mnew);
        float rs = p0 + p1;
        rs += __shfl_xor(rs, 8, 32);
        rs += __shfl_xor(rs, 4, 32);
        rs += __shfl_xor(rs, 2, 32);
        rs += __shfl_xor(rs, 1, 32);
        lrow[r] = lrow[r] * corr + rs;
        mrow[r] = mnew;
#pragma unroll
        for (int t16 = 0; t16 < 16; ++t16) o[t16][r] *= corr;

        const int mm = r + half * 8;
        plds[wid][mm][ln]      = (__bf16)p0;
        plds[wid][mm][ln + 16] = (__bf16)p1;
      }

      // O[16,256] += P[16,32] @ V[32,256-slice]; V rows key-contiguous in LDS
      v16bf ap = fragA_bf16(&plds[wid][ln][0], half);
#pragma unroll
      for (int t16 = 0; t16 < 16; ++t16) {
        const int d = dh * 256 + t16 * 16 + ln;
        o[t16] = wmma_bf16(ap, fragB_bf16(&vbuf[cur][d][0], half), o[t16]);
      }
    }

    async_wait_all();   // next tile landed in the other buffer
    __syncthreads();    // all waves done reading `cur` before it is reused
  }

  // normalize and store bf16 attention output [M, D]
#pragma unroll
  for (int r = 0; r < 8; ++r) {
    const float inv = 1.0f / lrow[r];
    const size_t rowo = ((size_t)bb * SEQ + qtb + r + half * 8) * DIM;
#pragma unroll
    for (int t16 = 0; t16 < 16; ++t16) {
      const int d = dh * 256 + t16 * 16 + ln;
      ao[rowo + d] = (__bf16)(o[t16][r] * inv);
    }
  }
}

// ---------------------------------------------------------------------------
// Kernel 5: output GEMM  out = AO @ Wo + residual(q), f32 store
// ---------------------------------------------------------------------------
__global__ void __launch_bounds__(256)
gemm_out512(const __bf16* __restrict__ Abf, const __bf16* __restrict__ Wt,
            const float* __restrict__ resid, float* __restrict__ out) {
  const int lane = threadIdx.x & 31, wid = threadIdx.x >> 5;
  const int half = lane >> 4, ln = lane & 15;
  const int m0 = blockIdx.x * 64 + (wid >> 1) * 16;
  const int n0 = blockIdx.y * 128 + (wid & 1) * 64;

  v8f acc[4];
#pragma unroll
  for (int t = 0; t < 4; ++t) acc[t] = (v8f){0, 0, 0, 0, 0, 0, 0, 0};

  const size_t arow = (size_t)(m0 + ln) * DIM;
  for (int ks = 0; ks < 16; ++ks) {
    const int kk = ks * 32;
    v16bf a = fragA_bf16(Abf + arow + kk, half);
#pragma unroll
    for (int t = 0; t < 4; ++t) {
      const __bf16* wrow = Wt + (size_t)(n0 + t * 16 + ln) * DIM + kk;
      acc[t] = wmma_bf16(a, fragB_bf16(wrow, half), acc[t]);
    }
  }

#pragma unroll
  for (int t = 0; t < 4; ++t) {
    const int n = n0 + t * 16 + ln;
#pragma unroll
    for (int r = 0; r < 8; ++r) {
      const int m = m0 + r + half * 8;
      out[(size_t)m * DIM + n] = acc[t][r] + resid[(size_t)m * DIM + n];
    }
  }
}

// ---------------------------------------------------------------------------
// Launch: prep -> 3 projection GEMMs -> flash attention -> output GEMM
// Workspace: 6 x 32MB bf16 tensors + 2MB weights ~= 194 MB from d_ws.
// ---------------------------------------------------------------------------
extern "C" void kernel_launch(void* const* d_in, const int* in_sizes, int n_in,
                              void* d_out, int out_size, void* d_ws,
                              size_t ws_size, hipStream_t stream) {
  const float* q    = (const float*)d_in[0];
  const float* k    = (const float*)d_in[1];
  const float* v    = (const float*)d_in[2];
  const float* pos  = (const float*)d_in[3];
  // d_in[4] = attention_mask (bool): unused, causal mask computed analytically
  const float* ln_g = (const float*)d_in[5];
  const float* ln_b = (const float*)d_in[6];
  const float* Wq   = (const float*)d_in[7];
  const float* Wk   = (const float*)d_in[8];
  const float* Wv   = (const float*)d_in[9];
  const float* pw1  = (const float*)d_in[10];
  const float* pb1  = (const float*)d_in[11];
  const float* pw2  = (const float*)d_in[12];
  const float* pb2  = (const float*)d_in[13];
  const float* Wo   = (const float*)d_in[14];
  float* out = (float*)d_out;

  char* ws = (char*)d_ws;
  size_t off = 0;
  auto carve = [&](size_t bytes) -> void* {
    void* p = ws + off;
    off += (bytes + 255) & ~(size_t)255;
    return p;
  };
  const size_t big = (size_t)MTOT * DIM * sizeof(__bf16);  // 32 MB
  __bf16* qn   = (__bf16*)carve(big);
  __bf16* peb  = (__bf16*)carve(big);
  __bf16* qpb  = (__bf16*)carve(big);
  __bf16* kpb  = (__bf16*)carve(big);
  __bf16* vpet = (__bf16*)carve(big);   // transposed [D, M]
  __bf16* aob  = (__bf16*)carve(big);
  __bf16* wqt  = (__bf16*)carve((size_t)DIM * DIM * 2);
  __bf16* wkt  = (__bf16*)carve((size_t)DIM * DIM * 2);
  __bf16* wvt  = (__bf16*)carve((size_t)DIM * DIM * 2);
  __bf16* wot  = (__bf16*)carve((size_t)DIM * DIM * 2);

  prep_w<<<4096, 256, 0, stream>>>(Wq, Wk, Wv, Wo, wqt, wkt, wvt, wot);
  prep_token<<<MTOT, 128, 0, stream>>>(q, pos, ln_g, ln_b, pw1, pb1, pw2, pb2,
                                       qn, peb);

  dim3 gg(MTOT / 64, DIM / 128);
  gemm512<0, 0, 0><<<gg, 256, 0, stream>>>(qn, nullptr, wqt, nullptr, qpb,
                                           nullptr);
  gemm512<1, 0, 0><<<gg, 256, 0, stream>>>(nullptr, k, wkt, nullptr, kpb,
                                           nullptr);
  gemm512<1, 1, 1><<<gg, 256, 0, stream>>>(nullptr, v, wvt, peb, nullptr,
                                           vpet);

  dim3 ga(SEQ / 64, BATCH);
  flash_attn<<<ga, 256, 0, stream>>>(qpb, kpb, vpet, aob);

  gemm_out512<<<gg, 256, 0, stream>>>(aob, wot, q, out);
}
